// TreeStructuredRCNNLSTM_31542239822276
// MI455X (gfx1250) — matrix operand order repaired and checked by reading, not compiled, not run on hardware
//
#include <hip/hip_runtime.h>

// MI455X (gfx1250, wave32) TreeStructuredRCNNLSTM.
// All large matmuls run through v_wmma_f32_16x16x32_bf16 (bf16 in, f32 acc).
// A-panels staged to LDS via async global->LDS loads (ASYNCcnt path);
// B fragments register-blocked over 2 M-tiles.

#define B_  64
#define S_  1024
#define E_  512
#define H_  512
#define NF_ 256

typedef unsigned int   u32;
typedef unsigned short u16;
typedef __attribute__((ext_vector_type(16))) __bf16 v16bf;
typedef __attribute__((ext_vector_type(8)))  float  v8f;
typedef __attribute__((ext_vector_type(4)))  u32    v4u;
typedef __attribute__((ext_vector_type(4)))  int    v4i;

#if __has_builtin(__builtin_amdgcn_global_load_async_to_lds_b128) && \
    __has_builtin(__builtin_amdgcn_s_wait_asynccnt)
#define USE_ASYNC_LDS 1
typedef __attribute__((address_space(1))) v4i as1_v4i;
typedef __attribute__((address_space(3))) v4i as3_v4i;
#else
#define USE_ASYNC_LDS 0
#endif

__device__ __forceinline__ void cp16_lds(u16* lds_dst, const u16* gsrc) {
#if USE_ASYNC_LDS
  __builtin_amdgcn_global_load_async_to_lds_b128((as1_v4i*)gsrc, (as3_v4i*)lds_dst,
                                                 0, 0);
#else
  *(v4u*)lds_dst = *(const v4u*)gsrc;
#endif
}
__device__ __forceinline__ void stage_fence() {
#if USE_ASYNC_LDS
  __builtin_amdgcn_s_wait_asynccnt(0);
#endif
}

__device__ __forceinline__ u16 f2bf(float f) {
  u32 u = __builtin_bit_cast(u32, f);
  u32 r = u + 0x7FFFu + ((u >> 16) & 1u);   // round-to-nearest-even
  return (u16)(r >> 16);
}
__device__ __forceinline__ float bf2f(u16 h) {
  u32 u = ((u32)h) << 16;
  return __builtin_bit_cast(float, u);
}
__device__ __forceinline__ float sigm(float x) { return 1.f / (1.f + __expf(-x)); }

union FragU { v4u u[2]; v16bf b; };

// A fragment (16x32 bf16): lane L(<16) row M=L holds K[0..7],K[16..23];
// lane L+16 same row, K offsets +8.  Two contiguous 16B loads.
__device__ __forceinline__ v16bf frag_a(const u16* base, int rs, int r0, int k0) {
  int lane = threadIdx.x & 31;
  const u16* p = base + (size_t)(r0 + (lane & 15)) * rs + k0 + 8 * (lane >> 4);
  FragU f;
  f.u[0] = *(const v4u*)p;
  f.u[1] = *(const v4u*)(p + 16);
  return f.b;
}
// B fragment (32x16 bf16, stored N-major [n][k]): lane L(<16) col N=L holds
// K[0..7],K[16..23]; lane L+16 same col, K offsets +8.  Two contiguous 16B loads.
__device__ __forceinline__ v16bf frag_b(const u16* base, int rs, int n0, int k0) {
  int lane = threadIdx.x & 31;
  const u16* p = base + (size_t)(n0 + (lane & 15)) * rs + k0 + 8 * (lane >> 4);
  FragU f;
  f.u[0] = *(const v4u*)p;
  f.u[1] = *(const v4u*)(p + 16);
  return f.b;
}

#define WMMA_BF16(a, b, c) \
  __builtin_amdgcn_wmma_f32_16x16x32_bf16(false, (a), false, (b), (short)0, (c), false, false)

#define V8F_ZERO {0.f, 0.f, 0.f, 0.f, 0.f, 0.f, 0.f, 0.f}

// ---------------------------------------------------------------- embedding
__global__ void embed_kernel(const int* __restrict__ x, const float* __restrict__ emb,
                             u16* __restrict__ e) {
  int bs = blockIdx.x;
  int tok = x[bs];
  const float4* src = (const float4*)(emb + (size_t)tok * E_);
  float4 v = src[threadIdx.x];                       // 128 threads x float4
  u32 p0 = (u32)f2bf(v.x) | ((u32)f2bf(v.y) << 16);
  u32 p1 = (u32)f2bf(v.z) | ((u32)f2bf(v.w) << 16);
  u32* d = (u32*)(e + (size_t)bs * E_);
  d[threadIdx.x * 2 + 0] = p0;
  d[threadIdx.x * 2 + 1] = p1;
}

// ------------------------------------------------------------ weight repack
__global__ void prep_bT(const float* __restrict__ src, u16* __restrict__ dst,
                        int K, int N) {
  int idx = blockIdx.x * 256 + threadIdx.x;
  if (idx >= N * K) return;
  int n = idx / K, k = idx - n * K;
  dst[idx] = f2bf(src[(size_t)k * N + n]);
}

__global__ void prep_wcomb(const float* __restrict__ Wt, u16* __restrict__ dst) {
  int idx = blockIdx.x * 256 + threadIdx.x;
  if (idx >= 2048 * 512) return;
  int n = idx / 512, k = idx - n * 512;
  float v = Wt[(size_t)k * 2048 + n] + 0.5f * Wt[(size_t)(512 + k) * 2048 + n];
  dst[idx] = f2bf(v);
}

__global__ void prep_conv(const float* __restrict__ w, u16* __restrict__ dst, int KW) {
  int idx = blockIdx.x * 256 + threadIdx.x;
  if (idx >= KW * NF_ * E_) return;
  int tap = idx / (NF_ * E_);
  int rem = idx - tap * NF_ * E_;
  int f = rem / E_, e = rem - f * E_;
  dst[idx] = f2bf(w[((size_t)f * E_ + e) * KW + tap]);
}

// -------------------------------------------------------- conv + relu + max
// grid (B, 8): each block handles 4 time-panels of 32 rows, 8 waves x 2
// N-tiles cover NF=256.  Partial maxima merged with float atomic-max
// (valid: relu >= 0, pooled pre-zeroed).
__global__ __launch_bounds__(256) void convpool_kernel(
    const u16* __restrict__ Ebf, const u16* __restrict__ Wc,
    const float* __restrict__ bias, float* __restrict__ pooled,
    int KW, int outOff) {
  __shared__ u16 As[36 * 528];
  int b = blockIdx.x;
  int tid = threadIdx.x, wave = tid >> 5, lane = tid & 31;
  int Tout = S_ - KW + 1;
  int n0a = wave * 32, n0b = n0a + 16;
  float run0 = 0.f, run1 = 0.f;
  const u16* Eb = Ebf + (size_t)b * S_ * E_;
  int R = 32 + KW - 1;

  for (int tt = blockIdx.y * 4; tt < blockIdx.y * 4 + 4; ++tt) {
    int t0 = tt * 32;
    for (int i = tid; i < R * 64; i += 256) {
      int r = i >> 6, ch = i & 63;
      int gr = t0 + r; if (gr > S_ - 1) gr = S_ - 1;
      cp16_lds(As + r * 528 + ch * 8, Eb + (size_t)gr * E_ + ch * 8);
    }
    stage_fence();
    __syncthreads();

    v8f acc[2][2] = {{V8F_ZERO, V8F_ZERO}, {V8F_ZERO, V8F_ZERO}};
    for (int tap = 0; tap < KW; ++tap) {
      const u16* Wtap = Wc + (size_t)tap * NF_ * E_;
      #pragma unroll
      for (int kc = 0; kc < E_ / 32; ++kc) {
        v16bf b0 = frag_b(Wtap, E_, n0a, kc * 32);
        v16bf b1 = frag_b(Wtap, E_, n0b, kc * 32);
        #pragma unroll
        for (int mt = 0; mt < 2; ++mt) {
          v16bf a = frag_a(As, 528, tap + mt * 16, kc * 32);
          acc[0][mt] = WMMA_BF16(a, b0, acc[0][mt]);
          acc[1][mt] = WMMA_BF16(a, b1, acc[1][mt]);
        }
      }
    }
    float bs0 = bias[n0a + (lane & 15)];
    float bs1 = bias[n0b + (lane & 15)];
    #pragma unroll
    for (int mt = 0; mt < 2; ++mt) {
      #pragma unroll
      for (int r = 0; r < 8; ++r) {
        int t = t0 + mt * 16 + r + 8 * (lane >> 4);
        if (t < Tout) {
          run0 = fmaxf(run0, fmaxf(acc[0][mt][r] + bs0, 0.f));
          run1 = fmaxf(run1, fmaxf(acc[1][mt][r] + bs1, 0.f));
        }
      }
    }
    __syncthreads();
  }
  run0 = fmaxf(run0, __shfl_xor(run0, 16, 32));
  run1 = fmaxf(run1, __shfl_xor(run1, 16, 32));
  if (lane < 16) {
    float* p0 = &pooled[(size_t)b * (3 * NF_) + outOff + n0a + lane];
    float* p1 = &pooled[(size_t)b * (3 * NF_) + outOff + n0b + lane];
    __hip_atomic_fetch_max(p0, run0, __ATOMIC_RELAXED, __HIP_MEMORY_SCOPE_AGENT);
    __hip_atomic_fetch_max(p1, run1, __ATOMIC_RELAXED, __HIP_MEMORY_SCOPE_AGENT);
  }
}

// ---------------------------------------------- cnn_proj[b] = cnn[b]@Wl[512:]+bl
__global__ void cnnproj_kernel(const float* __restrict__ pooled,
                               const float* __restrict__ Wl,
                               const float* __restrict__ bl,
                               float* __restrict__ proj) {
  int b = blockIdx.x;
  int j = blockIdx.y * 256 + threadIdx.x;
  float s = bl[j];
  const float* pb = pooled + (size_t)b * (3 * NF_);
  for (int k = 0; k < 3 * NF_; ++k)
    s += pb[k] * Wl[(size_t)(E_ + k) * H_ + j];
  proj[(size_t)b * H_ + j] = s;
}

// --------------------------------------- leaf = e @ Wl[:512] + proj[b] (bf16)
// 512 threads / 16 waves, M=32 per block; each B fragment feeds 2 WMMAs.
__global__ __launch_bounds__(512) void leaf_kernel(
    const u16* __restrict__ Ebf, const u16* __restrict__ WlT,
    const float* __restrict__ proj, u16* __restrict__ leaf) {
  __shared__ u16 As[32 * 528];
  int r0 = blockIdx.x * 32;
  int tid = threadIdx.x, wave = tid >> 5, lane = tid & 31;
  for (int i = tid; i < 32 * 64; i += 512) {
    int r = i >> 6, ch = i & 63;
    cp16_lds(As + r * 528 + ch * 8, Ebf + (size_t)(r0 + r) * E_ + ch * 8);
  }
  stage_fence();
  __syncthreads();
  v8f z = V8F_ZERO;
  v8f acc[4] = {z, z, z, z};                 // [q*2 + mt]
  for (int kc = 0; kc < E_ / 32; ++kc) {
    v16bf a0 = frag_a(As, 528, 0, kc * 32);
    v16bf a1 = frag_a(As, 528, 16, kc * 32);
    #pragma unroll
    for (int q = 0; q < 2; ++q) {
      v16bf bb = frag_b(WlT, H_, wave * 32 + q * 16, kc * 32);
      acc[q * 2 + 0] = WMMA_BF16(a0, bb, acc[q * 2 + 0]);
      acc[q * 2 + 1] = WMMA_BF16(a1, bb, acc[q * 2 + 1]);
    }
  }
  int b = r0 / S_;                            // S_ % 32 == 0: single batch
  #pragma unroll
  for (int q = 0; q < 2; ++q) {
    int j = wave * 32 + q * 16 + (lane & 15);
    float cp = proj[(size_t)b * H_ + j];
    #pragma unroll
    for (int mt = 0; mt < 2; ++mt)
      #pragma unroll
      for (int r = 0; r < 8; ++r) {
        int row = r0 + mt * 16 + r + 8 * (lane >> 4);
        leaf[(size_t)row * H_ + j] = f2bf(acc[q * 2 + mt][r] + cp);
      }
  }
}

// --------------------------- fused LSTM cell: g = A@W + bt ; gates ; h,c out
// 512 threads / 16 waves, M=32. Wave w owns gate-matched columns
// [w*32, w*32+32) so sigmoid/tanh combine stays in registers.
__global__ __launch_bounds__(512) void cell_kernel(
    const u16* __restrict__ A, const u16* __restrict__ W,
    const u16* __restrict__ Csum, const float* __restrict__ bt,
    u16* __restrict__ Hout, u16* __restrict__ Cout) {
  __shared__ u16 As[32 * 528];
  int r0 = blockIdx.x * 32;
  int tid = threadIdx.x, wave = tid >> 5, lane = tid & 31;
  for (int i = tid; i < 32 * 64; i += 512) {
    int r = i >> 6, ch = i & 63;
    cp16_lds(As + r * 528 + ch * 8, A + (size_t)(r0 + r) * H_ + ch * 8);
  }
  stage_fence();
  __syncthreads();
  v8f z = V8F_ZERO;
  v8f acc[16];                                // [(gg*2+q)*2 + mt]
  #pragma unroll
  for (int i = 0; i < 16; ++i) acc[i] = z;
  for (int kc = 0; kc < H_ / 32; ++kc) {
    v16bf a0 = frag_a(As, 528, 0, kc * 32);
    v16bf a1 = frag_a(As, 528, 16, kc * 32);
    #pragma unroll
    for (int gg = 0; gg < 4; ++gg) {
      #pragma unroll
      for (int q = 0; q < 2; ++q) {
        int n0 = gg * 512 + wave * 32 + q * 16;
        v16bf bb = frag_b(W, H_, n0, kc * 32);
        int idx = (gg * 2 + q) * 2;
        acc[idx + 0] = WMMA_BF16(a0, bb, acc[idx + 0]);
        acc[idx + 1] = WMMA_BF16(a1, bb, acc[idx + 1]);
      }
    }
  }
  #pragma unroll
  for (int q = 0; q < 2; ++q) {
    int j = wave * 32 + q * 16 + (lane & 15);
    float bF = bt[j], bI = bt[512 + j], bC = bt[1024 + j], bO = bt[1536 + j];
    #pragma unroll
    for (int mt = 0; mt < 2; ++mt) {
      v8f F  = acc[(0 * 2 + q) * 2 + mt];
      v8f I  = acc[(1 * 2 + q) * 2 + mt];
      v8f CT = acc[(2 * 2 + q) * 2 + mt];
      v8f O  = acc[(3 * 2 + q) * 2 + mt];
      #pragma unroll
      for (int r = 0; r < 8; ++r) {
        int row = r0 + mt * 16 + r + 8 * (lane >> 4);
        float cs = Csum ? bf2f(Csum[(size_t)row * H_ + j]) : 0.f;
        float c = sigm(F[r] + bF) * cs + sigm(I[r] + bI) * tanhf(CT[r] + bC);
        float h = sigm(O[r] + bO) * tanhf(c);
        Hout[(size_t)row * H_ + j] = f2bf(h);
        Cout[(size_t)row * H_ + j] = f2bf(c);
      }
    }
  }
}

// ------------------------------------------------------------- pairwise sum
__global__ void pairsum_kernel(const u16* __restrict__ h, const u16* __restrict__ c,
                               u16* __restrict__ hs, u16* __restrict__ cs, int half) {
  int orow = blockIdx.x;
  int b = orow / half, i = orow - b * half;
  size_t rA = ((size_t)b * 2 * half + 2 * i) * H_;
  size_t rB = rA + H_;
  int t = threadIdx.x;
  u32 ha = ((const u32*)(h + rA))[t], hb = ((const u32*)(h + rB))[t];
  u32 ca = ((const u32*)(c + rA))[t], cb = ((const u32*)(c + rB))[t];
  float h0 = bf2f((u16)(ha & 0xFFFF)) + bf2f((u16)(hb & 0xFFFF));
  float h1 = bf2f((u16)(ha >> 16))    + bf2f((u16)(hb >> 16));
  float c0 = bf2f((u16)(ca & 0xFFFF)) + bf2f((u16)(cb & 0xFFFF));
  float c1 = bf2f((u16)(ca >> 16))    + bf2f((u16)(cb >> 16));
  ((u32*)(hs + (size_t)orow * H_))[t] = (u32)f2bf(h0) | ((u32)f2bf(h1) << 16);
  ((u32*)(cs + (size_t)orow * H_))[t] = (u32)f2bf(c0) | ((u32)f2bf(c1) << 16);
}

// -------------------------------------------------------------- final MLP
__global__ void mlp_kernel(const u16* __restrict__ root, const float* __restrict__ W1,
                           const float* __restrict__ b1, const float* __restrict__ W2,
                           const float* __restrict__ b2, float* __restrict__ out) {
  __shared__ float hid[256];
  int b = blockIdx.x, j = threadIdx.x;
  float s = b1[j];
  const u16* rb = root + (size_t)b * H_;
  for (int k = 0; k < H_; ++k)
    s += bf2f(rb[k]) * W1[(size_t)k * 256 + j];
  hid[j] = fmaxf(s, 0.f);
  __syncthreads();
  if (j < 3) {
    float o = b2[j];
    for (int k = 0; k < 256; ++k) o += hid[k] * W2[k * 3 + j];
    out[b * 3 + j] = o;
  }
}

// ===========================================================================
extern "C" void kernel_launch(void* const* d_in, const int* in_sizes, int n_in,
                              void* d_out, int out_size, void* d_ws, size_t ws_size,
                              hipStream_t stream) {
  const int*   x   = (const int*)  d_in[0];
  const float* emb = (const float*)d_in[1];
  const float* w3  = (const float*)d_in[2];
  const float* b3  = (const float*)d_in[3];
  const float* w4  = (const float*)d_in[4];
  const float* b4  = (const float*)d_in[5];
  const float* w5  = (const float*)d_in[6];
  const float* b5  = (const float*)d_in[7];
  const float* Wl  = (const float*)d_in[8];
  const float* bl  = (const float*)d_in[9];
  const float* Wt  = (const float*)d_in[10];
  const float* bt  = (const float*)d_in[11];
  const float* W1  = (const float*)d_in[12];
  const float* b1  = (const float*)d_in[13];
  const float* W2  = (const float*)d_in[14];
  const float* b2  = (const float*)d_in[15];
  float* out = (float*)d_out;

  char* ws = (char*)d_ws;
  size_t off = 0;
  auto walloc = [&](size_t bytes) -> void* {
    void* p = ws + off;
    off = (off + bytes + 255) & ~(size_t)255;
    return p;
  };

  u16*   e_bf   = (u16*)walloc((size_t)B_ * S_ * E_ * 2);         //  64 MB
  u16*   leaf   = (u16*)walloc((size_t)B_ * S_ * H_ * 2);         //  64 MB
  u16*   hA     = (u16*)walloc((size_t)131072 * H_ * 2);          // 128 MB arena
  u16*   cA     = (u16*)walloc((size_t)131072 * H_ * 2);          // 128 MB arena
  u16*   hs     = (u16*)walloc((size_t)(B_ * S_ / 2) * H_ * 2);   //  32 MB
  u16*   cs     = (u16*)walloc((size_t)(B_ * S_ / 2) * H_ * 2);   //  32 MB
  u16*   wlT    = (u16*)walloc((size_t)H_ * E_ * 2);
  u16*   wx     = (u16*)walloc((size_t)2048 * 512 * 2);
  u16*   wcomb  = (u16*)walloc((size_t)2048 * 512 * 2);
  u16*   w3t    = (u16*)walloc((size_t)3 * NF_ * E_ * 2);
  u16*   w4t    = (u16*)walloc((size_t)4 * NF_ * E_ * 2);
  u16*   w5t    = (u16*)walloc((size_t)5 * NF_ * E_ * 2);
  float* pooled = (float*)walloc((size_t)B_ * 3 * NF_ * 4);
  float* proj   = (float*)walloc((size_t)B_ * H_ * 4);

  // ---- stage 0: gather + weight conversion/repack
  embed_kernel<<<B_ * S_, 128, 0, stream>>>(x, emb, e_bf);
  prep_bT<<<(512 * 512 + 255) / 256, 256, 0, stream>>>(Wl, wlT, 512, 512);
  prep_bT<<<(512 * 2048 + 255) / 256, 256, 0, stream>>>(Wt + (size_t)512 * 2048, wx, 512, 2048);
  prep_wcomb<<<(512 * 2048 + 255) / 256, 256, 0, stream>>>(Wt, wcomb);
  prep_conv<<<(3 * NF_ * E_ + 255) / 256, 256, 0, stream>>>(w3, w3t, 3);
  prep_conv<<<(4 * NF_ * E_ + 255) / 256, 256, 0, stream>>>(w4, w4t, 4);
  prep_conv<<<(5 * NF_ * E_ + 255) / 256, 256, 0, stream>>>(w5, w5t, 5);

  // ---- stage 1: conv banks (WMMA) + max-pool (atomic-max combine)
  (void)hipMemsetAsync(pooled, 0, (size_t)B_ * 3 * NF_ * 4, stream);
  convpool_kernel<<<dim3(B_, 8), 256, 0, stream>>>(e_bf, w3t, b3, pooled, 3, 0);
  convpool_kernel<<<dim3(B_, 8), 256, 0, stream>>>(e_bf, w4t, b4, pooled, 4, 256);
  convpool_kernel<<<dim3(B_, 8), 256, 0, stream>>>(e_bf, w5t, b5, pooled, 5, 512);
  cnnproj_kernel<<<dim3(B_, 2), 256, 0, stream>>>(pooled, Wl, bl, proj);

  // ---- stage 2: leaf projection (WMMA)
  leaf_kernel<<<B_ * S_ / 32, 512, 0, stream>>>(e_bf, wlT, proj, leaf);

  // ---- stage 3: tree LSTM. Level 0: h_sum = 0 -> g = leaf @ Wt_x + bt
  cell_kernel<<<B_ * S_ / 32, 512, 0, stream>>>(leaf, wx, nullptr, bt, hA, cA);
  size_t rowsOff = 0;
  int n = S_;
  while (n > 1) {
    int half = n >> 1;
    pairsum_kernel<<<B_ * half, 256, 0, stream>>>(hA + rowsOff * H_, cA + rowsOff * H_,
                                                  hs, cs, half);
    size_t nxt = rowsOff + (size_t)B_ * n;
    cell_kernel<<<B_ * half / 32, 512, 0, stream>>>(hs, wcomb, cs, bt,
                                                    hA + nxt * H_, cA + nxt * H_);
    rowsOff = nxt;
    n = half;
  }

  // ---- stage 4: root MLP
  mlp_kernel<<<B_, 256, 0, stream>>>(hA + rowsOff * H_, W1, b1, W2, b2, out);
}
